// AttnBlock_58205396795667
// MI455X (gfx1250) — matrix operand order repaired and checked by reading, not compiled, not run on hardware
//
#include <hip/hip_runtime.h>
#include <hip/hip_bf16.h>

typedef __attribute__((ext_vector_type(16))) _Float16 v16h;
typedef __attribute__((ext_vector_type(8)))  _Float16 v8h;
typedef __attribute__((ext_vector_type(8)))  float    v8f;

#define WMMA(A, B, C) \
  __builtin_amdgcn_wmma_f32_16x16x32_f16(false, (A), false, (B), (short)0, (C), false, false)

static __device__ __forceinline__ v16h cat8(v8h lo, v8h hi) {
  v16h r;
#pragma unroll
  for (int i = 0; i < 8; ++i) { r[i] = lo[i]; r[i + 8] = hi[i]; }
  return r;
}

// ---------------------------------------------------------------------------
// Kernel 1: GroupNorm (32 groups, 2 ch/group) + transpose [B,C,N] -> [B,N,C] f16
// grid = B*32 groups, block = 256
// ---------------------------------------------------------------------------
__global__ void gn_kernel(const float* __restrict__ x, const float* __restrict__ gamma,
                          const float* __restrict__ beta, _Float16* __restrict__ h_t) {
  const int bg = blockIdx.x;            // b*32 + g
  const int b = bg >> 5, g = bg & 31;
  const float* xp = x + ((size_t)(b * 64 + g * 2)) * 4096;   // 8192 floats
  float s = 0.f, ss = 0.f;
  for (int i = threadIdx.x; i < 8192; i += 256) {
    float v = xp[i]; s += v; ss += v * v;
  }
  __shared__ float bs[256], bss[256];
  bs[threadIdx.x] = s; bss[threadIdx.x] = ss;
  __syncthreads();
  for (int off = 128; off > 0; off >>= 1) {
    if ((int)threadIdx.x < off) {
      bs[threadIdx.x] += bs[threadIdx.x + off];
      bss[threadIdx.x] += bss[threadIdx.x + off];
    }
    __syncthreads();
  }
  const float mean = bs[0] * (1.f / 8192.f);
  const float var  = bss[0] * (1.f / 8192.f) - mean * mean;
  const float rstd = rsqrtf(var + 1e-5f);
  const float g0 = gamma[g * 2], b0 = beta[g * 2];
  const float g1 = gamma[g * 2 + 1], b1 = beta[g * 2 + 1];
  for (int i = threadIdx.x; i < 8192; i += 256) {
    const int cl = i >> 12, n = i & 4095;
    float v = (xp[i] - mean) * rstd;
    v = v * (cl ? g1 : g0) + (cl ? b1 : b0);
    h_t[((size_t)(b * 4096 + n)) * 64 + g * 2 + cl] = (_Float16)v;
  }
}

// ---------------------------------------------------------------------------
// Kernel 2: Q/K/V 1x1 conv via WMMA. q,k -> [B,N,C] f16 ; v -> vt [B,C,N] f16
// grid = B*256 row tiles (16 rows), block = 128 (4 waves; wave handles (mat,cg))
// ---------------------------------------------------------------------------
__global__ void qkv_kernel(const _Float16* __restrict__ h,
                           const float* __restrict__ wq, const float* __restrict__ bq,
                           const float* __restrict__ wk, const float* __restrict__ bk,
                           const float* __restrict__ wv, const float* __restrict__ bv,
                           _Float16* __restrict__ q, _Float16* __restrict__ k,
                           _Float16* __restrict__ vt) {
  const int tile = blockIdx.x;          // 0..1023
  const int b = tile >> 8;
  const int n0 = (tile & 255) << 4;
  const int lane = threadIdx.x & 31;
  const int wave = threadIdx.x >> 5;
  const int col = lane & 15, hi = lane >> 4;

  // A = h tile [16 x 64], K-split c0 = 0..31, c1 = 32..63
  const _Float16* arow = h + ((size_t)(b * 4096 + n0 + col)) * 64;
  const v16h A0 = *(const v16h*)(arow + hi * 16);
  const v16h A1 = *(const v16h*)(arow + 32 + hi * 16);

  const float* Ws[3] = {wq, wk, wv};
  const float* Bi[3] = {bq, bk, bv};

  for (int p = wave; p < 12; p += 4) {
    const int mat = p >> 2, cg = p & 3;
    const int ch = cg * 16 + col;
    // B[k=ci, n=co] = W[co][ci]; lane needs 16 contiguous ci at hi*16 (+32)
    const float* w0 = Ws[mat] + ch * 64 + hi * 16;
    v16h B0, B1;
#pragma unroll
    for (int t = 0; t < 16; ++t) { B0[t] = (_Float16)w0[t]; B1[t] = (_Float16)w0[32 + t]; }
    v8f acc = {};
    acc = WMMA(A0, B0, acc);
    acc = WMMA(A1, B1, acc);
    const float bias = Bi[mat][ch];
    if (mat < 2) {
      _Float16* dst = (mat == 0 ? q : k);
#pragma unroll
      for (int j = 0; j < 8; ++j)
        dst[((size_t)(b * 4096 + n0 + hi * 8 + j)) * 64 + ch] = (_Float16)(acc[j] + bias);
    } else {
      v8h pk;
#pragma unroll
      for (int j = 0; j < 8; ++j) pk[j] = (_Float16)(acc[j] + bias);
      *(v8h*)(vt + ((size_t)(b * 64 + ch)) * 4096 + n0 + hi * 8) = pk;
    }
  }
}

// ---------------------------------------------------------------------------
// Kernel 3: flash attention. One wave = 16 queries, streams all 4096 keys in
// 32-key tiles. S^T = K_tile @ Q^T (keys = rows) so softmax reduces in-lane;
// the two 16-key S^T accumulators map exactly onto one A-fragment for P.V.
// grid = 256, block = 128 (4 independent waves)
// ---------------------------------------------------------------------------
__global__ void __launch_bounds__(128)
attn_kernel(const _Float16* __restrict__ q, const _Float16* __restrict__ k,
            const _Float16* __restrict__ vt, _Float16* __restrict__ o) {
  const int wave = threadIdx.x >> 5;
  const int tile = blockIdx.x * 4 + wave;     // 0..1023
  const int b = tile >> 8;
  const int q0 = (tile & 255) << 4;
  const int lane = threadIdx.x & 31;
  const int col = lane & 15, hi = lane >> 4;
  const int hi8 = hi * 8;

  // B operand = Q^T, loaded once: lane = query col, channels hi8..+7 and 16+hi8..+7 (+32 for Bq1)
  const _Float16* qrow = q + ((size_t)(b * 4096 + q0 + col)) * 64;
  const v16h Bq0 = cat8(*(const v8h*)(qrow + hi8),      *(const v8h*)(qrow + 16 + hi8));
  const v16h Bq1 = cat8(*(const v8h*)(qrow + 32 + hi8), *(const v8h*)(qrow + 48 + hi8));

  const _Float16* kbase = k + ((size_t)b * 4096) * 64;
  const _Float16* vbase = vt + ((size_t)b * 64) * 4096;

  v8f O0 = {}, O1 = {}, O2 = {}, O3 = {};
  float m_run = -3.0e38f, l_run = 0.f;
  const float sc = 0.125f * 1.44269504088896f;   // 1/sqrt(64) * log2(e)

  for (int kb = 0; kb < 4096; kb += 32) {
    // A = K rows (16 keys x 64 ch), two row tiles x two K(channel) chunks
    const _Float16* kr0 = kbase + ((size_t)(kb + col)) * 64 + hi8;
    const _Float16* kr1 = kbase + ((size_t)(kb + 16 + col)) * 64 + hi8;
    const v16h Ak00 = cat8(*(const v8h*)(kr0),      *(const v8h*)(kr0 + 16));
    const v16h Ak01 = cat8(*(const v8h*)(kr0 + 32), *(const v8h*)(kr0 + 48));
    const v16h Ak10 = cat8(*(const v8h*)(kr1),      *(const v8h*)(kr1 + 16));
    const v16h Ak11 = cat8(*(const v8h*)(kr1 + 32), *(const v8h*)(kr1 + 48));

    v8f s0 = {}, s1 = {};
    s0 = WMMA(Ak00, Bq0, s0); s0 = WMMA(Ak01, Bq1, s0);   // keys kb..kb+15
    s1 = WMMA(Ak10, Bq0, s1); s1 = WMMA(Ak11, Bq1, s1);   // keys kb+16..kb+31

    // online softmax over keys: lane holds one query (col), 16 key values
    float mx = -3.0e38f;
#pragma unroll
    for (int j = 0; j < 8; ++j) {
      s0[j] *= sc; s1[j] *= sc;
      mx = fmaxf(mx, fmaxf(s0[j], s1[j]));
    }
    mx = fmaxf(mx, __shfl_xor(mx, 16, 32));
    const float m_new = fmaxf(m_run, mx);
    const float alpha = exp2f(m_run - m_new);
    m_run = m_new;

    v16h Ap;                     // P in A-fragment layout, no lane exchange needed
    float rs = 0.f;
#pragma unroll
    for (int j = 0; j < 8; ++j) {
      const float p0 = exp2f(s0[j] - m_new);
      const float p1 = exp2f(s1[j] - m_new);
      rs += p0 + p1;
      Ap[j] = (_Float16)p0;
      Ap[j + 8] = (_Float16)p1;
    }
    rs += __shfl_xor(rs, 16, 32);
    l_run = l_run * alpha + rs;

    // rescale O (rows = queries hi8+j): broadcast alpha from lane hi8+j
#pragma unroll
    for (int j = 0; j < 8; ++j) {
      const float aj = __shfl(alpha, hi8 + j, 32);
      O0[j] *= aj; O1[j] *= aj; O2[j] *= aj; O3[j] *= aj;
    }

    // B operand = V [32 keys x 16 ch]; vt channel-major -> keys hi8..+7 and 16+hi8..+7 per lane
    const _Float16* vr = vbase + kb + hi8;
    const v16h Bv0 = cat8(*(const v8h*)(vr + ((size_t)(col)) * 4096), *(const v8h*)(vr + ((size_t)(col)) * 4096 + 16));
    const v16h Bv1 = cat8(*(const v8h*)(vr + ((size_t)(16 + col)) * 4096), *(const v8h*)(vr + ((size_t)(16 + col)) * 4096 + 16));
    const v16h Bv2 = cat8(*(const v8h*)(vr + ((size_t)(32 + col)) * 4096), *(const v8h*)(vr + ((size_t)(32 + col)) * 4096 + 16));
    const v16h Bv3 = cat8(*(const v8h*)(vr + ((size_t)(48 + col)) * 4096), *(const v8h*)(vr + ((size_t)(48 + col)) * 4096 + 16));

    O0 = WMMA(Ap, Bv0, O0);
    O1 = WMMA(Ap, Bv1, O1);
    O2 = WMMA(Ap, Bv2, O2);
    O3 = WMMA(Ap, Bv3, O3);
  }

  const float invl = 1.f / l_run;
#pragma unroll
  for (int j = 0; j < 8; ++j) {
    const float rj = __shfl(invl, hi8 + j, 32);
    _Float16* orow = o + ((size_t)(b * 4096 + q0 + hi8 + j)) * 64 + col;
    orow[0]  = (_Float16)(O0[j] * rj);
    orow[16] = (_Float16)(O1[j] * rj);
    orow[32] = (_Float16)(O2[j] * rj);
    orow[48] = (_Float16)(O3[j] * rj);
  }
}

// ---------------------------------------------------------------------------
// Kernel 4: output proj via WMMA + bias + residual, write f32 [B,C,H,W]
// grid = B*256 row tiles, block = 128 (wave = channel group)
// ---------------------------------------------------------------------------
__global__ void proj_kernel(const _Float16* __restrict__ a, const float* __restrict__ wp,
                            const float* __restrict__ bp, const float* __restrict__ x,
                            float* __restrict__ out) {
  const int tile = blockIdx.x;          // 0..1023
  const int b = tile >> 8;
  const int n0 = (tile & 255) << 4;
  const int cg = threadIdx.x >> 5;
  const int lane = threadIdx.x & 31;
  const int col = lane & 15, hi = lane >> 4;

  const _Float16* arow = a + ((size_t)(b * 4096 + n0 + col)) * 64;
  const v16h A0 = *(const v16h*)(arow + hi * 16);
  const v16h A1 = *(const v16h*)(arow + 32 + hi * 16);

  const int ch = cg * 16 + col;
  const float* w0 = wp + ch * 64 + hi * 16;
  v16h B0, B1;
#pragma unroll
  for (int t = 0; t < 16; ++t) { B0[t] = (_Float16)w0[t]; B1[t] = (_Float16)w0[32 + t]; }

  v8f acc = {};
  acc = WMMA(A0, B0, acc);
  acc = WMMA(A1, B1, acc);

  const float bias = bp[ch];
  const size_t base = ((size_t)(b * 64 + ch)) * 4096 + n0 + hi * 8;
#pragma unroll
  for (int j = 0; j < 8; ++j)
    out[base + j] = x[base + j] + acc[j] + bias;
}

// ---------------------------------------------------------------------------
extern "C" void kernel_launch(void* const* d_in, const int* in_sizes, int n_in,
                              void* d_out, int out_size, void* d_ws, size_t ws_size,
                              hipStream_t stream) {
  const float* x  = (const float*)d_in[0];
  const float* gg = (const float*)d_in[1];
  const float* gb = (const float*)d_in[2];
  const float* wq = (const float*)d_in[3];
  const float* bq = (const float*)d_in[4];
  const float* wk = (const float*)d_in[5];
  const float* bk = (const float*)d_in[6];
  const float* wv = (const float*)d_in[7];
  const float* bv = (const float*)d_in[8];
  const float* wp = (const float*)d_in[9];
  const float* bp = (const float*)d_in[10];
  float* out = (float*)d_out;

  // Workspace: 5 x (B*N*C = 1<<20) f16 buffers = 10 MB
  _Float16* h_t  = (_Float16*)d_ws;
  _Float16* qb   = h_t + (1 << 20);
  _Float16* kb   = qb  + (1 << 20);
  _Float16* vt   = kb  + (1 << 20);
  _Float16* at   = vt  + (1 << 20);

  gn_kernel  <<<128,  256, 0, stream>>>(x, gg, gb, h_t);
  qkv_kernel <<<1024, 128, 0, stream>>>(h_t, wq, bq, wk, bk, wv, bv, qb, kb, vt);
  attn_kernel<<<256,  128, 0, stream>>>(qb, kb, vt, at);
  proj_kernel<<<1024, 128, 0, stream>>>(at, wp, bp, x, out);
}
